// Pts_attention_72954314490251
// MI455X (gfx1250) — matrix operand order repaired and checked by reading, not compiled, not run on hardware
//
#include <hip/hip_runtime.h>

#define C_DIM 256
#define N_DIM 8192
#define M_DIM 2048
#define B_DIM 4
#define NT 128   // N rows per workgroup
#define MT 64    // M tile in attention loop

typedef __attribute__((ext_vector_type(16))) __bf16         v16bf;
typedef __attribute__((ext_vector_type(8)))  float          v8f;
typedef __attribute__((ext_vector_type(4)))  unsigned int   u32x4;
typedef __attribute__((ext_vector_type(4)))  float          f32x4;
typedef __attribute__((ext_vector_type(4)))  unsigned short u16x4;

union FragU { u32x4 q[2]; v16bf v; unsigned short s[16]; };

// LDS layout constants (attention kernel)
#define P_BYTES   (8 * 16 * 72 * 2)            // per-wave P scratch: 18432 B
#define B_STRIDE  264                           // b tile row stride (bf16)
#define C_STRIDE  72                            // cT tile row stride (bf16)
#define B_BYTES   (64 * B_STRIDE * 2)           // 33792 B
#define C_BYTES   (256 * C_STRIDE * 2)          // 36864 B
#define BUF_BYTES (B_BYTES + C_BYTES)           // 70656 B
#define SMEM_ATTN (P_BYTES + 2 * BUF_BYTES)     // 159744 B (O staging aliases bufs)

// log2(e)/sqrt(C): folded into fc1 output so softmax runs in base-2.
#define SCL_LOG2E 0.0901684400f

static __device__ __forceinline__ unsigned short f2bf(float f) {
  unsigned int u = __builtin_bit_cast(unsigned int, f);
  u += 0x7FFFu + ((u >> 16) & 1u);          // round-to-nearest-even
  return (unsigned short)(u >> 16);
}
static __device__ __forceinline__ void bf16st(unsigned short* p, float f) { *p = f2bf(f); }

static __device__ __forceinline__ v8f wmma_bf16(v16bf a, v16bf b, v8f c) {
  return __builtin_amdgcn_wmma_f32_16x16x32_bf16(false, a, false, b, (short)0, c,
                                                 false, false);
}

static __device__ __forceinline__ void ld_frag(FragU& f, const unsigned short* p) {
  f.q[0] = *(const u32x4*)(p);
  f.q[1] = *(const u32x4*)(p + 8);
}

// Async DMA of one 16B chunk per lane: global -> LDS (tracked by ASYNCcnt).
static __device__ __forceinline__ void async_ld16(const unsigned short* lds,
                                                  const unsigned short* gaddr) {
  unsigned int loff = (unsigned int)(size_t)lds;  // flat LDS addr: low 32 = LDS offset
  asm volatile("global_load_async_to_lds_b128 %0, %1, off"
               :: "v"(loff), "v"(gaddr) : "memory");
}
static __device__ __forceinline__ void wait_async0() {
  asm volatile("s_wait_asynccnt 0" ::: "memory");
}

// ---------------------------------------------------------------------------
// One-shot f32 -> bf16 weight conversion (W is reused by every workgroup;
// convert once so fc inner loops are pure 32B bf16 loads).
// ---------------------------------------------------------------------------
__global__ __launch_bounds__(256) void cvt_kernel(const float* __restrict__ src,
                                                  unsigned short* __restrict__ dst,
                                                  int n4) {
  int i = (blockIdx.x * 256 + threadIdx.x) * 4;
  if (i < n4 * 4) {
    f32x4 v = *(const f32x4*)(src + i);
    u16x4 o;
    o.x = f2bf(v.x); o.y = f2bf(v.y); o.z = f2bf(v.z); o.w = f2bf(v.w);
    *(u16x4*)(dst + i) = o;
  }
}

// ---------------------------------------------------------------------------
// fc kernel: out = (xin^T @ W^T + bias) * oscale, bf16 output.
//   xin: [B, C, rows] f32 ; Wbf: [C, C] bf16 ; bias: [C] f32
//   TR=false: out [B, rows, C] ; TR=true: out [B, C, rows]
// ---------------------------------------------------------------------------
template <bool TR>
__global__ __launch_bounds__(256) void fc_kernel(const float* __restrict__ xin,
                                                 const unsigned short* __restrict__ Wbf,
                                                 const float* __restrict__ bias,
                                                 unsigned short* __restrict__ outp,
                                                 int rows, float oscale) {
  extern __shared__ char smem[];
  unsigned short* ldsX = (unsigned short*)smem;  // [128][264] bf16 (padded rows)

  const int tid  = threadIdx.x;
  const int lane = tid & 31;
  const int wave = tid >> 5;
  const int h    = lane >> 4;   // wave half (0: lanes 0-15, 1: lanes 16-31)
  const int l16  = lane & 15;
  const int n0   = blockIdx.x * NT;
  const int bz   = blockIdx.y;

  // Stage x tile into LDS, transposed to [n_local][c], f32 -> bf16.
  const float* xB = xin + (size_t)bz * C_DIM * rows;
  for (int idx = tid; idx < NT * C_DIM; idx += 256) {
    int nl = idx & (NT - 1);
    int c  = idx >> 7;
    ldsX[nl * 264 + c] = f2bf(xB[(size_t)c * rows + n0 + nl]);
  }
  __syncthreads();

  // Preload the 8 A-fragments (K = 0..255) for this wave's 16 rows.
  v16bf af[8];
  {
    const unsigned short* ar = ldsX + (wave * 16 + l16) * 264;
    #pragma unroll
    for (int k = 0; k < 8; ++k) {
      FragU f;
      f.q[0] = *(const u32x4*)(ar + k * 32 + h * 16);
      f.q[1] = *(const u32x4*)(ar + k * 32 + h * 16 + 8);
      af[k] = f.v;
    }
  }

  unsigned short* oB = outp + (size_t)bz * rows * C_DIM;

  #pragma unroll 1
  for (int cj = 0; cj < 16; ++cj) {          // output column tile (16 c_out)
    v8f acc = {};
    // B-frag of W^T: lane holds column c_out = cj*16+l16, 16 contiguous c_in.
    const unsigned short* wrow = Wbf + (size_t)(cj * 16 + l16) * C_DIM + h * 16;
    FragU f0, f1;
    ld_frag(f0, wrow);
    ld_frag(f1, wrow + 32);
    #pragma unroll
    for (int k = 0; k < 8; ++k) {
      FragU& cur = (k & 1) ? f1 : f0;
      acc = wmma_bf16(af[k], cur.v, acc);
      if (k + 2 < 8) ld_frag(cur, wrow + (k + 2) * 32);
    }
    float bv = bias[cj * 16 + l16];
    #pragma unroll
    for (int v = 0; v < 8; ++v) {
      float r = (acc[v] + bv) * oscale;
      int ng = n0 + wave * 16 + v + 8 * h;   // D layout: row = v + 8*half
      int cg = cj * 16 + l16;                // D layout: col = lane%16
      if (TR) bf16st(&oB[(size_t)cg * rows + ng], r);
      else    bf16st(&oB[(size_t)ng * C_DIM + cg], r);
    }
  }
}

// ---------------------------------------------------------------------------
// Cooperative async staging of one M-tile: b [64][256] and cT [256][64] bf16.
// 2048 + 2048 16B chunks spread over 256 threads = 8 + 8 async ops/thread.
// ---------------------------------------------------------------------------
static __device__ __forceinline__ void stage_tiles(const unsigned short* bkB,
                                                   const unsigned short* cTB,
                                                   int m0,
                                                   unsigned short* bBuf,
                                                   unsigned short* cBuf,
                                                   int tid) {
  #pragma unroll
  for (int t = 0; t < 8; ++t) {
    int cid = t * 256 + tid;          // b tile: row = cid/32, 16B chunk = cid%32
    int row = cid >> 5, col = cid & 31;
    async_ld16(bBuf + row * B_STRIDE + col * 8,
               bkB + (size_t)(m0 + row) * C_DIM + col * 8);
  }
  #pragma unroll
  for (int t = 0; t < 8; ++t) {
    int cid = t * 256 + tid;          // cT tile: c-row = cid/8, chunk = cid%8
    int crow = cid >> 3, col = cid & 7;
    async_ld16(cBuf + crow * C_STRIDE + col * 8,
               cTB + (size_t)crow * M_DIM + m0 + col * 8);
  }
}

// ---------------------------------------------------------------------------
// Fused flash attention (scores pre-scaled by log2e/sqrt(C) in fc1):
//   S2 = a @ b^T (base-2 domain), online softmax over M via exp2,
//   O = softmax(S) @ c, out = x + O^T
// ---------------------------------------------------------------------------
__global__ __launch_bounds__(256) void attn_kernel(const float* __restrict__ x,
                                                   const unsigned short* __restrict__ aq,
                                                   const unsigned short* __restrict__ bk,
                                                   const unsigned short* __restrict__ cT,
                                                   float* __restrict__ out) {
  extern __shared__ char smem[];
  unsigned short* ldsP  = (unsigned short*)smem;                // 8*[16][72] bf16
  unsigned short* tile0 = (unsigned short*)(smem + P_BYTES);    // double buffers
  unsigned short* tile1 = (unsigned short*)(smem + P_BYTES + BUF_BYTES);
  float*          ldsO  = (float*)(smem + P_BYTES);             // aliases tile bufs

  const int tid  = threadIdx.x;
  const int lane = tid & 31;
  const int wave = tid >> 5;
  const int h    = lane >> 4;
  const int l16  = lane & 15;
  const int n0   = blockIdx.x * NT;
  const int bz   = blockIdx.y;

  const unsigned short* aqB = aq + (size_t)bz * N_DIM * C_DIM;
  const unsigned short* bkB = bk + (size_t)bz * M_DIM * C_DIM;
  const unsigned short* cTB = cT + (size_t)bz * C_DIM * M_DIM;

  // Kick off tile 0 staging immediately.
  stage_tiles(bkB, cTB, 0, tile0, tile0 + B_BYTES / 2, tid);

  // Preload Q fragments (this wave's 16 rows, full K=256): 8 frags.
  v16bf qf[8];
  {
    const unsigned short* qrow = aqB + (size_t)(n0 + wave * 16 + l16) * C_DIM;
    #pragma unroll
    for (int k = 0; k < 8; ++k) {
      FragU f;
      f.q[0] = *(const u32x4*)(qrow + k * 32 + h * 16);
      f.q[1] = *(const u32x4*)(qrow + k * 32 + h * 16 + 8);
      qf[k] = f.v;
    }
  }

  v8f o[16];
  #pragma unroll
  for (int j = 0; j < 16; ++j) o[j] = {};
  float mrow[8], lrow[8];
  #pragma unroll
  for (int v = 0; v < 8; ++v) { mrow[v] = -3.0e38f; lrow[v] = 0.0f; }

  unsigned short* ldsPw = ldsP + wave * 16 * 72;

  #pragma unroll 1
  for (int it = 0; it < M_DIM / MT; ++it) {
    const int m0 = it * MT;
    unsigned short* bCur = (it & 1) ? tile1 : tile0;
    unsigned short* cCur = bCur + B_BYTES / 2;

    // Wait for this tile's DMA (issued last iteration), make it WG-visible,
    // then immediately start DMAing the next tile into the other buffer.
    wait_async0();
    __syncthreads();
    if (it + 1 < M_DIM / MT) {
      unsigned short* bNxt = (it & 1) ? tile0 : tile1;
      stage_tiles(bkB, cTB, m0 + MT, bNxt, bNxt + B_BYTES / 2, tid);
    }

    // ---- S = Q @ b^T for this wave's 16 rows x 64 m-cols (double-buffered) --
    v8f s[4];
    #pragma unroll
    for (int mi = 0; mi < 4; ++mi) {
      const unsigned short* brow = bCur + (mi * 16 + l16) * B_STRIDE + h * 16;
      v8f acc = {};
      FragU f0, f1;
      ld_frag(f0, brow);
      ld_frag(f1, brow + 32);
      #pragma unroll
      for (int k = 0; k < 8; ++k) {
        FragU& cur = (k & 1) ? f1 : f0;
        acc = wmma_bf16(qf[k], cur.v, acc);
        if (k + 2 < 8) ld_frag(cur, brow + (k + 2) * 32);
      }
      s[mi] = acc;
    }

    // ---- per-row max of this tile (lane-local rows v + 8*h) ----
    float tmax[8];
    #pragma unroll
    for (int v = 0; v < 8; ++v) {
      float t = fmaxf(fmaxf(s[0][v], s[1][v]), fmaxf(s[2][v], s[3][v]));
      t = fmaxf(t, __shfl_xor(t, 1, 32));
      t = fmaxf(t, __shfl_xor(t, 2, 32));
      t = fmaxf(t, __shfl_xor(t, 4, 32));
      t = fmaxf(t, __shfl_xor(t, 8, 32));   // masks < 16: stays within half
      tmax[v] = t;
    }

    // ---- online softmax rescale (base-2 domain) ----
    float esum[8];
    #pragma unroll
    for (int v = 0; v < 8; ++v) {
      float mnew = fmaxf(mrow[v], tmax[v]);
      float corr = __builtin_amdgcn_exp2f(mrow[v] - mnew);
      mrow[v] = mnew;
      lrow[v] *= corr;
      #pragma unroll
      for (int j = 0; j < 16; ++j) o[j][v] *= corr;
      esum[v] = 0.0f;
    }

    // ---- P = exp2(S - m), spill to per-wave LDS in row-major [16][64] ----
    #pragma unroll
    for (int mi = 0; mi < 4; ++mi) {
      #pragma unroll
      for (int v = 0; v < 8; ++v) {
        float p = __builtin_amdgcn_exp2f(s[mi][v] - mrow[v]);
        esum[v] += p;
        ldsPw[(v + 8 * h) * 72 + mi * 16 + l16] = f2bf(p);
      }
    }
    #pragma unroll
    for (int v = 0; v < 8; ++v) {
      float t = esum[v];
      t += __shfl_xor(t, 1, 32);
      t += __shfl_xor(t, 2, 32);
      t += __shfl_xor(t, 4, 32);
      t += __shfl_xor(t, 8, 32);
      lrow[v] += t;
    }

    // ---- reload P in A-fragment layout (same-wave LDS ops are in-order) ----
    v16bf pf[2];
    #pragma unroll
    for (int kk = 0; kk < 2; ++kk) {
      const unsigned short* pr = ldsPw + l16 * 72 + kk * 32 + h * 16;
      FragU f;
      f.q[0] = *(const u32x4*)(pr);
      f.q[1] = *(const u32x4*)(pr + 8);
      pf[kk] = f.v;
    }

    // ---- O += P @ c (32 pipelined steps over the cT tile in LDS) ----
    {
      const unsigned short* cbase = cCur + l16 * C_STRIDE + h * 16;
      FragU g0, g1;
      ld_frag(g0, cbase);
      ld_frag(g1, cbase + 32);
      #pragma unroll
      for (int st = 0; st < 32; ++st) {
        FragU& cur = (st & 1) ? g1 : g0;
        o[st >> 1] = wmma_bf16(pf[st & 1], cur.v, o[st >> 1]);
        if (st + 2 < 32) {
          int nst = st + 2;
          ld_frag(cur, cbase + (nst >> 1) * 16 * C_STRIDE + (nst & 1) * 32);
        }
      }
    }
  }

  // ---- normalize and stage O^T through LDS (aliases dead tile buffers) ----
  #pragma unroll
  for (int v = 0; v < 8; ++v) {
    float rl = 1.0f / lrow[v];
    #pragma unroll
    for (int j = 0; j < 16; ++j) o[j][v] *= rl;
  }
  __syncthreads();   // everyone done reading the last tile before aliasing
  #pragma unroll
  for (int j = 0; j < 16; ++j) {
    int cg = j * 16 + l16;
    #pragma unroll
    for (int v = 0; v < 8; ++v) {
      int nl = wave * 16 + v + 8 * h;
      ldsO[cg * 132 + nl] = o[j][v];
    }
  }
  __syncthreads();

  const float* xB = x + (size_t)bz * C_DIM * N_DIM;
  float*       oG = out + (size_t)bz * C_DIM * N_DIM;
  for (int idx = tid; idx < C_DIM * NT; idx += 256) {
    int c  = idx >> 7;
    int nl = idx & (NT - 1);
    size_t g = (size_t)c * N_DIM + n0 + nl;
    oG[g] = xB[g] + ldsO[c * 132 + nl];
  }
}

// ---------------------------------------------------------------------------
extern "C" void kernel_launch(void* const* d_in, const int* in_sizes, int n_in,
                              void* d_out, int out_size, void* d_ws, size_t ws_size,
                              hipStream_t stream) {
  (void)in_sizes; (void)n_in; (void)out_size; (void)ws_size;
  const float* x    = (const float*)d_in[0];
  const float* x_pt = (const float*)d_in[1];
  const float* W1   = (const float*)d_in[2];
  const float* b1   = (const float*)d_in[3];
  const float* W2   = (const float*)d_in[4];
  const float* b2   = (const float*)d_in[5];
  const float* W3   = (const float*)d_in[6];
  const float* b3   = (const float*)d_in[7];
  float* out = (float*)d_out;

  unsigned short* aw  = (unsigned short*)d_ws;                // a  [B,N,C] bf16
  unsigned short* bw  = aw + (size_t)B_DIM * N_DIM * C_DIM;   // b  [B,M,C] bf16
  unsigned short* cw  = bw + (size_t)B_DIM * M_DIM * C_DIM;   // c^T[B,C,M] bf16
  unsigned short* w1b = cw + (size_t)B_DIM * M_DIM * C_DIM;   // W bf16 copies
  unsigned short* w2b = w1b + C_DIM * C_DIM;
  unsigned short* w3b = w2b + C_DIM * C_DIM;

  const int wn4 = C_DIM * C_DIM / 4;
  cvt_kernel<<<dim3(wn4 / 256), 256, 0, stream>>>(W1, w1b, wn4);
  cvt_kernel<<<dim3(wn4 / 256), 256, 0, stream>>>(W2, w2b, wn4);
  cvt_kernel<<<dim3(wn4 / 256), 256, 0, stream>>>(W3, w3b, wn4);

  size_t fcSh = (size_t)NT * 264 * 2;
  fc_kernel<false><<<dim3(N_DIM / NT, B_DIM), 256, fcSh, stream>>>(x,    w1b, b1, aw, N_DIM,
                                                                   SCL_LOG2E);
  fc_kernel<false><<<dim3(M_DIM / NT, B_DIM), 256, fcSh, stream>>>(x_pt, w2b, b2, bw, M_DIM,
                                                                   1.0f);
  fc_kernel<true ><<<dim3(M_DIM / NT, B_DIM), 256, fcSh, stream>>>(x_pt, w3b, b3, cw, M_DIM,
                                                                   1.0f);

  attn_kernel<<<dim3(N_DIM / NT, B_DIM), 256, SMEM_ATTN, stream>>>(x, aw, bw, cw, out);
}
